// _GGPLCGRTMLP_45165876085583
// MI455X (gfx1250) — matrix operand order repaired and checked by reading, not compiled, not run on hardware
//
#include <hip/hip_runtime.h>

// ---------------------------------------------------------------------------
// Model constants (from the reference)
// ---------------------------------------------------------------------------
#define B_    16384
#define D_    512
#define NB_   8
#define E_    4
#define DT_   256
#define GH_   256
#define SH_   128
#define TS_   16
#define SPD_  272
#define SPH_  68
#define CFD_  278
#define CFH_  69

// GEMM tiling: 256 threads = 8 waves; each wave owns 16 rows x 128 cols
#define BM 128
#define BN 128
#define BK 32
#define LDA 40   // padded K stride for A tile (elements)  -> 80B rows, 16B aligned
#define LDB 40   // padded K stride for transposed B tile

// CDNA5 async global->LDS copy (ASYNCcnt-tracked), guarded so either toolchain compiles.
// Toolchain signature (from diagnostics): (global int4*, local int4*, imm offset, imm cpol)
#if defined(__HIP_DEVICE_COMPILE__) && __has_builtin(__builtin_amdgcn_global_load_async_to_lds_b128)
#define USE_ASYNC_COPY 1
typedef __attribute__((__vector_size__(16))) int v4i_t;
typedef __attribute__((address_space(1))) v4i_t* g_b128_t;
typedef __attribute__((address_space(3))) v4i_t* l_b128_t;
__device__ inline void wait_asynccnt0() {
#if __has_builtin(__builtin_amdgcn_s_wait_asynccnt)
  __builtin_amdgcn_s_wait_asynccnt(0);
#else
  asm volatile("s_wait_asynccnt 0" ::: "memory");
#endif
}
#else
#define USE_ASYNC_COPY 0
#endif

typedef __attribute__((ext_vector_type(16))) __bf16 bf16x16;
typedef __attribute__((ext_vector_type(8)))  float  f32x8;

union FragBF { uint4 q[2]; bf16x16 v; };

__device__ inline float wave_sum(float v) {
#pragma unroll
  for (int m = 16; m >= 1; m >>= 1) v += __shfl_xor(v, m, 32);
  return v;
}
__device__ inline float wave_max(float v) {
#pragma unroll
  for (int m = 16; m >= 1; m >>= 1) v = fmaxf(v, __shfl_xor(v, m, 32));
  return v;
}

// ---------------------------------------------------------------------------
// Prep kernels
// ---------------------------------------------------------------------------
__global__ void bps_kernel(const float* __restrict__ bp_min,
                           const float* __restrict__ delta,
                           float* __restrict__ bps) {
  int d = blockIdx.x * blockDim.x + threadIdx.x;
  if (d < D_) {
    float acc = bp_min[d];
#pragma unroll
    for (int j = 0; j < NB_; ++j) {
      float t = delta[d * NB_ + j];
      float sp = (t > 20.f) ? t : log1pf(expf(t));
      acc += sp;
      bps[d * NB_ + j] = acc;
    }
  }
}

__global__ void cvt_kernel(const float* __restrict__ src, __bf16* __restrict__ dst, int n) {
  int i = blockIdx.x * blockDim.x + threadIdx.x;
  int stride = gridDim.x * blockDim.x;
  for (; i < n; i += stride) dst[i] = (__bf16)src[i];
}

// ---------------------------------------------------------------------------
// Fused GGPL embedding + projection GEMM:
//   x_in = x_num + 0.1*(flat(emb) @ proj_W + proj_b)
// A[16384 x 2048] generated on the fly (8 features x E=4 per 32-wide K step).
// ---------------------------------------------------------------------------
__global__ __launch_bounds__(256) void emb_proj_kernel(
    const float* __restrict__ x_num, const float* __restrict__ basis_w,
    const float* __restrict__ basis_b, const float* __restrict__ bps,
    const __bf16* __restrict__ Wb, const float* __restrict__ proj_b,
    float* __restrict__ xinF, __bf16* __restrict__ xinB) {
  __shared__ __align__(16) __bf16 Atile[BM * LDA];
  __shared__ __align__(16) __bf16 Btile[BN * LDB];
  const int tid = threadIdx.x;
  const int wv = tid >> 5, lane = tid & 31;
  const int hl = lane >> 4, l16 = lane & 15;
  const int m0 = blockIdx.x * BM, n0 = blockIdx.y * BN;

  f32x8 acc[8] = {};

  for (int s = 0; s < (D_ * E_) / BK; ++s) {          // 64 K-steps of 32
    const int dbase = s * 8;
    // --- generate A tile: 128 rows x 32 (8 features x 4 embed dims) ---
#pragma unroll
    for (int i = 0; i < 4; ++i) {
      int pid = tid + i * 256;                        // 0..1023
      int r = pid >> 3, df = pid & 7;
      int d = dbase + df;
      float x = x_num[(size_t)(m0 + r) * D_ + d];
      float bb[NB_ + 1];
      bb[0] = x;
#pragma unroll
      for (int j = 0; j < NB_; ++j) bb[j + 1] = fmaxf(x - bps[d * NB_ + j], 0.f);
#pragma unroll
      for (int e = 0; e < E_; ++e) {
        float a = basis_b[d * E_ + e];
#pragma unroll
        for (int j = 0; j < NB_ + 1; ++j) a += bb[j] * basis_w[d * 36 + j * E_ + e];
        Atile[r * LDA + df * E_ + e] = (__bf16)a;
      }
    }
    // --- stage proj_W tile transposed: Btile[n][k] ---
#pragma unroll
    for (int i = 0; i < 2; ++i) {
      int cid = tid + i * 256;
      int kk = cid >> 4, nq = cid & 15;
      uint4 w = *(const uint4*)(Wb + (size_t)(s * BK + kk) * D_ + n0 + nq * 8);
      const __bf16* wsrc = (const __bf16*)&w;
#pragma unroll
      for (int t = 0; t < 8; ++t) Btile[(nq * 8 + t) * LDB + kk] = wsrc[t];
    }
    __syncthreads();
    // --- fragments per ISA layout; hoist all B frags, then 8 WMMAs ---
    FragBF a, b[8];
    a.q[0] = *(const uint4*)&Atile[(wv * 16 + l16) * LDA + hl * 8];
    a.q[1] = *(const uint4*)&Atile[(wv * 16 + l16) * LDA + 16 + hl * 8];
#pragma unroll
    for (int cg = 0; cg < 8; ++cg) {
      b[cg].q[0] = *(const uint4*)&Btile[(cg * 16 + l16) * LDB + hl * 8];
      b[cg].q[1] = *(const uint4*)&Btile[(cg * 16 + l16) * LDB + 16 + hl * 8];
    }
#pragma unroll
    for (int cg = 0; cg < 8; ++cg)
      acc[cg] = __builtin_amdgcn_wmma_f32_16x16x32_bf16(false, a.v, false, b[cg].v,
                                                        (short)0, acc[cg], false, false);
    __syncthreads();
  }
  // --- epilogue: x_in = x_num + 0.1*(acc + proj_b); store f32 + bf16 ---
#pragma unroll
  for (int cg = 0; cg < 8; ++cg) {
    int col = n0 + cg * 16 + l16;
    float pb = proj_b[col];
#pragma unroll
    for (int v = 0; v < 8; ++v) {
      int row = m0 + wv * 16 + hl * 8 + v;
      size_t idx = (size_t)row * D_ + col;
      float xi = x_num[idx] + 0.1f * (acc[cg][v] + pb);
      xinF[idx] = xi;
      xinB[idx] = (__bf16)xi;
    }
  }
}

// ---------------------------------------------------------------------------
// Generic bf16 WMMA GEMM, M fixed by grid.x*BM. Epilogue modes:
//   0: outB = bf16(relu(acc+bias))
//   1: g = sigmoid(acc+bias) -> outF ; outB = bf16(auxF * g)      (gate layer)
//   2: outF = relu(acc+bias)                                       (backbone h)
// ---------------------------------------------------------------------------
template <int MODE>
__global__ __launch_bounds__(256) void gemm_bf16_kernel(
    const __bf16* __restrict__ A, const __bf16* __restrict__ W,
    const float* __restrict__ bias, const float* __restrict__ auxF,
    float* __restrict__ outF, __bf16* __restrict__ outB, int N, int K) {
  __shared__ __align__(16) __bf16 Atile[BM * LDA];
  __shared__ __align__(16) __bf16 Btile[BN * LDB];
  const int tid = threadIdx.x;
  const int wv = tid >> 5, lane = tid & 31;
  const int hl = lane >> 4, l16 = lane & 15;
  const int m0 = blockIdx.x * BM, n0 = blockIdx.y * BN;

  f32x8 acc[8] = {};

  for (int k0 = 0; k0 < K; k0 += BK) {
    // --- stage A tile (straight copy): async global->LDS when available ---
#if USE_ASYNC_COPY
#pragma unroll
    for (int i = 0; i < 2; ++i) {
      int cid = tid + i * 256;
      int r = cid >> 2, kq = cid & 3;
      __builtin_amdgcn_global_load_async_to_lds_b128(
          (g_b128_t)(void*)(A + (size_t)(m0 + r) * K + k0 + kq * 8),
          (l_b128_t)(void*)&Atile[r * LDA + kq * 8], 0, 0);
    }
#else
#pragma unroll
    for (int i = 0; i < 2; ++i) {
      int cid = tid + i * 256;
      int r = cid >> 2, kq = cid & 3;
      *(uint4*)&Atile[r * LDA + kq * 8] =
          *(const uint4*)(A + (size_t)(m0 + r) * K + k0 + kq * 8);
    }
    if (k0 + BK < K) {  // gfx1250 global_prefetch_b8 for next A slab
      int r = tid >> 2, kq = tid & 3;
      __builtin_prefetch(A + (size_t)(m0 + r) * K + k0 + BK + kq * 8, 0, 1);
    }
#endif
    // --- stage W tile transposed while the A copy is in flight ---
#pragma unroll
    for (int i = 0; i < 2; ++i) {
      int cid = tid + i * 256;
      int kk = cid >> 4, nq = cid & 15;
      uint4 w = *(const uint4*)(W + (size_t)(k0 + kk) * N + n0 + nq * 8);
      const __bf16* wsrc = (const __bf16*)&w;
#pragma unroll
      for (int t = 0; t < 8; ++t) Btile[(nq * 8 + t) * LDB + kk] = wsrc[t];
    }
#if USE_ASYNC_COPY
    wait_asynccnt0();
#endif
    __syncthreads();
    // --- hoist all fragment loads, then issue the 8-WMMA chain ---
    FragBF a, b[8];
    a.q[0] = *(const uint4*)&Atile[(wv * 16 + l16) * LDA + hl * 8];
    a.q[1] = *(const uint4*)&Atile[(wv * 16 + l16) * LDA + 16 + hl * 8];
#pragma unroll
    for (int cg = 0; cg < 8; ++cg) {
      b[cg].q[0] = *(const uint4*)&Btile[(cg * 16 + l16) * LDB + hl * 8];
      b[cg].q[1] = *(const uint4*)&Btile[(cg * 16 + l16) * LDB + 16 + hl * 8];
    }
#pragma unroll
    for (int cg = 0; cg < 8; ++cg)
      acc[cg] = __builtin_amdgcn_wmma_f32_16x16x32_bf16(false, a.v, false, b[cg].v,
                                                        (short)0, acc[cg], false, false);
    __syncthreads();
  }
#pragma unroll
  for (int cg = 0; cg < 8; ++cg) {
    int col = n0 + cg * 16 + l16;
    float bs = bias[col];
#pragma unroll
    for (int v = 0; v < 8; ++v) {
      int row = m0 + wv * 16 + hl * 8 + v;
      size_t idx = (size_t)row * N + col;
      float val = acc[cg][v] + bs;
      if (MODE == 0) {
        outB[idx] = (__bf16)fmaxf(val, 0.f);
      } else if (MODE == 1) {
        float gate = 1.f / (1.f + __expf(-val));
        outF[idx] = gate;
        outB[idx] = (__bf16)(auxF[idx] * gate);
      } else {
        outF[idx] = fmaxf(val, 0.f);
      }
    }
  }
}

// ---------------------------------------------------------------------------
// Heads: one wave32 per row. Gate stats, top-8, token summary, safe/spec/conf
// LayerNorm MLP heads, final blend.
// ---------------------------------------------------------------------------
__global__ __launch_bounds__(256) void heads_kernel(
    const float* __restrict__ g, const float* __restrict__ x_in, const float* __restrict__ h,
    const float* __restrict__ base_W, const float* __restrict__ base_b,
    const float* __restrict__ sl_g, const float* __restrict__ sl_b,
    const float* __restrict__ sW1, const float* __restrict__ sb1,
    const float* __restrict__ sW2, const float* __restrict__ sb2,
    const float* __restrict__ tW1, const float* __restrict__ tb1,
    const float* __restrict__ tW2, const float* __restrict__ tb2,
    const float* __restrict__ pl_g, const float* __restrict__ pl_b,
    const float* __restrict__ pW1, const float* __restrict__ pb1,
    const float* __restrict__ pW2, const float* __restrict__ pb2,
    const float* __restrict__ cl_g, const float* __restrict__ cl_b,
    const float* __restrict__ cW1, const float* __restrict__ cb1,
    const float* __restrict__ cW2, const float* __restrict__ cb2,
    float* __restrict__ out) {
  __shared__ float hrow[8][DT_];
  __shared__ float sbuf[8][SPD_];
  __shared__ float cbuf[8][280];
  const int wv = threadIdx.x >> 5, lane = threadIdx.x & 31;
  const int row = blockIdx.x * 8 + wv;
  const float NEG = -3.0e38f;

  // ---- gate stats + top-k(8) over 512 gates (16 values per lane) ----
  float gv[16], xv[16];
#pragma unroll
  for (int j = 0; j < 16; ++j) {
    int i = j * 32 + lane;
    gv[j] = g[(size_t)row * D_ + i];
    xv[j] = x_in[(size_t)row * D_ + i];
  }
  float s = 0.f, ss = 0.f, mx = NEG;
#pragma unroll
  for (int j = 0; j < 16; ++j) { s += gv[j]; ss += gv[j] * gv[j]; mx = fmaxf(mx, gv[j]); }
  s = wave_sum(s); ss = wave_sum(ss); mx = wave_max(mx);
  float gmean = s * (1.f / D_);
  float gstd = sqrtf(fmaxf(ss * (1.f / D_) - gmean * gmean, 0.f));

  float cur[16];
#pragma unroll
  for (int j = 0; j < 16; ++j) cur[j] = gv[j];
  float tg[8], tx[8];
#pragma unroll
  for (int k = 0; k < 8; ++k) {
    float bv = NEG; int bi = 0x7fffffff;
#pragma unroll
    for (int j = 0; j < 16; ++j) { int idx = j * 32 + lane; if (cur[j] > bv) { bv = cur[j]; bi = idx; } }
#pragma unroll
    for (int m = 16; m >= 1; m >>= 1) {
      float ov = __shfl_xor(bv, m, 32);
      int oi = __shfl_xor(bi, m, 32);
      if (ov > bv || (ov == bv && oi < bi)) { bv = ov; bi = oi; }
    }
    int owner = bi & 31, slot = bi >> 5;
    float xc = 0.f;
#pragma unroll
    for (int j = 0; j < 16; ++j)
      if (j == slot) { xc = xv[j]; if (lane == owner) cur[j] = NEG; }
    tg[k] = bv;
    tx[k] = __shfl(xc, owner, 32);
  }
  float wsum = 1e-6f;
#pragma unroll
  for (int k = 0; k < 8; ++k) wsum += tg[k];

  // ---- token summary z_top (lanes 0..15 hold one TS dim each) ----
  float z = 0.f;
  if (lane < TS_) {
#pragma unroll
    for (int k = 0; k < 8; ++k) {
      float xw = tx[k] * (tg[k] / wsum);
      float tok = tb2[lane];
#pragma unroll
      for (int t = 0; t < TS_; ++t) {
        float hd = fmaxf(xw * tW1[t] + tb1[t], 0.f);
        tok += hd * tW2[t * TS_ + lane];
      }
      z += tok;
    }
    z *= 0.125f;
  }

  // ---- h row, y_base, LN(h) stats ----
  float hv8[8];
#pragma unroll
  for (int j = 0; j < 8; ++j) {
    float v = h[(size_t)row * DT_ + j * 32 + lane];
    hv8[j] = v;
    hrow[wv][j * 32 + lane] = v;
  }
  float yb = 0.f, hs = 0.f, hss = 0.f;
#pragma unroll
  for (int j = 0; j < 8; ++j) {
    yb += hv8[j] * base_W[j * 32 + lane];
    hs += hv8[j]; hss += hv8[j] * hv8[j];
  }
  yb = wave_sum(yb) + base_b[0];
  hs = wave_sum(hs); hss = wave_sum(hss);
  float hm = hs * (1.f / DT_);
  float hinv = rsqrtf(fmaxf(hss * (1.f / DT_) - hm * hm, 0.f) + 1e-5f);

  // ---- safe head: normalize into sbuf, 256->128->1 ----
#pragma unroll
  for (int j = 0; j < 8; ++j) {
    int i = j * 32 + lane;
    sbuf[wv][i] = (hv8[j] - hm) * hinv * sl_g[i] + sl_b[i];
  }
  __syncthreads();
  float dsp = 0.f;
#pragma unroll
  for (int q = 0; q < 4; ++q) {
    int u = lane + 32 * q;
    float hu = sb1[u];
    for (int i = 0; i < DT_; ++i) hu += sbuf[wv][i] * sW1[(size_t)i * SH_ + u];
    dsp += fmaxf(hu, 0.f) * sW2[u];
  }
  float dsafe = wave_sum(dsp) + sb2[0];
  __syncthreads();

  // ---- spec head: concat(h, z_top) [272] -> LN -> 68 -> 1 ----
#pragma unroll
  for (int j = 0; j < 8; ++j) sbuf[wv][j * 32 + lane] = hv8[j];
  if (lane < TS_) sbuf[wv][DT_ + lane] = z;
  __syncthreads();
  float ps = 0.f, pss = 0.f;
#pragma unroll
  for (int j = 0; j < 9; ++j) {
    int i = j * 32 + lane;
    if (i < SPD_) { float v = sbuf[wv][i]; ps += v; pss += v * v; }
  }
  ps = wave_sum(ps); pss = wave_sum(pss);
  float pm = ps * (1.f / SPD_);
  float pinv = rsqrtf(fmaxf(pss * (1.f / SPD_) - pm * pm, 0.f) + 1e-5f);
#pragma unroll
  for (int j = 0; j < 9; ++j) {
    int i = j * 32 + lane;
    if (i < SPD_) sbuf[wv][i] = (sbuf[wv][i] - pm) * pinv * pl_g[i] + pl_b[i];
  }
  __syncthreads();
  float psp = 0.f;
#pragma unroll
  for (int q = 0; q < 3; ++q) {
    int u = lane + 32 * q;
    if (u < SPH_) {
      float hu = pb1[u];
      for (int i = 0; i < SPD_; ++i) hu += sbuf[wv][i] * pW1[(size_t)i * SPH_ + u];
      psp += fmaxf(hu, 0.f) * pW2[u];
    }
  }
  float dspec = wave_sum(psp) + pb2[0];

  // ---- conf gate: [h, yb, gmean, gmax, gstd, z, |dsafe|, |dspec|] [278] ----
#pragma unroll
  for (int j = 0; j < 8; ++j) cbuf[wv][j * 32 + lane] = hv8[j];
  if (lane == 0) {
    cbuf[wv][256] = yb; cbuf[wv][257] = gmean; cbuf[wv][258] = mx; cbuf[wv][259] = gstd;
    cbuf[wv][276] = fabsf(dsafe); cbuf[wv][277] = fabsf(dspec);
  }
  if (lane < TS_) cbuf[wv][260 + lane] = z;
  __syncthreads();
  float cs = 0.f, css = 0.f;
#pragma unroll
  for (int j = 0; j < 9; ++j) {
    int i = j * 32 + lane;
    if (i < CFD_) { float v = cbuf[wv][i]; cs += v; css += v * v; }
  }
  cs = wave_sum(cs); css = wave_sum(css);
  float cm = cs * (1.f / CFD_);
  float cinv = rsqrtf(fmaxf(css * (1.f / CFD_) - cm * cm, 0.f) + 1e-5f);
#pragma unroll
  for (int j = 0; j < 9; ++j) {
    int i = j * 32 + lane;
    if (i < CFD_) cbuf[wv][i] = (cbuf[wv][i] - cm) * cinv * cl_g[i] + cl_b[i];
  }
  __syncthreads();
  float csp = 0.f;
#pragma unroll
  for (int q = 0; q < 3; ++q) {
    int u = lane + 32 * q;
    if (u < CFH_) {
      float hu = cb1[u];
      for (int i = 0; i < CFD_; ++i) hu += cbuf[wv][i] * cW1[(size_t)i * CFH_ + u];
      csp += fmaxf(hu, 0.f) * cW2[u];
    }
  }
  float clog = wave_sum(csp) + cb2[0];
  float gamma = 1.f / (1.f + __expf(-clog));

  if (lane == 0) out[row] = yb + dsafe + gamma * dspec;
}

// ---------------------------------------------------------------------------
// Workspace layout (bytes)
// ---------------------------------------------------------------------------
static constexpr size_t O_BPS   = 0;                                   // 512*8*4
static constexpr size_t O_PWB   = 16384;                               // 2048*512 bf16
static constexpr size_t O_GW1B  = O_PWB  + 2048ull * 512 * 2;
static constexpr size_t O_GW2B  = O_GW1B + 512ull * 256 * 2;
static constexpr size_t O_BW1B  = O_GW2B + 256ull * 512 * 2;
static constexpr size_t O_BW2B  = O_BW1B + 512ull * 256 * 2;
static constexpr size_t O_XINF  = O_BW2B + 256ull * 256 * 2;           // x_in f32
static constexpr size_t O_GF    = O_XINF + (size_t)B_ * D_ * 4;        // g f32
static constexpr size_t O_HF    = O_GF   + (size_t)B_ * D_ * 4;        // h f32
static constexpr size_t O_XBF   = O_HF   + (size_t)B_ * DT_ * 4;       // x_in/x_gated bf16 (reused)
static constexpr size_t O_HIDBF = O_XBF  + (size_t)B_ * D_ * 2;        // hidden bf16 (reused)

extern "C" void kernel_launch(void* const* d_in, const int* in_sizes, int n_in,
                              void* d_out, int out_size, void* d_ws, size_t ws_size,
                              hipStream_t stream) {
  const float* x_num     = (const float*)d_in[0];
  const float* basis_w   = (const float*)d_in[1];
  const float* basis_b   = (const float*)d_in[2];
  const float* bp_min    = (const float*)d_in[3];
  const float* bp_dr     = (const float*)d_in[4];
  const float* proj_W    = (const float*)d_in[5];
  const float* proj_b    = (const float*)d_in[6];
  const float* gate_W1   = (const float*)d_in[7];
  const float* gate_b1   = (const float*)d_in[8];
  const float* gate_W2   = (const float*)d_in[9];
  const float* gate_b2   = (const float*)d_in[10];
  const float* bb_W1     = (const float*)d_in[11];
  const float* bb_b1     = (const float*)d_in[12];
  const float* bb_W2     = (const float*)d_in[13];
  const float* bb_b2     = (const float*)d_in[14];
  const float* base_W    = (const float*)d_in[15];
  const float* base_b    = (const float*)d_in[16];
  const float* safe_ln_g = (const float*)d_in[17];
  const float* safe_ln_b = (const float*)d_in[18];
  const float* safe_W1   = (const float*)d_in[19];
  const float* safe_b1   = (const float*)d_in[20];
  const float* safe_W2   = (const float*)d_in[21];
  const float* safe_b2   = (const float*)d_in[22];
  const float* top_W1    = (const float*)d_in[23];
  const float* top_b1    = (const float*)d_in[24];
  const float* top_W2    = (const float*)d_in[25];
  const float* top_b2    = (const float*)d_in[26];
  const float* spec_ln_g = (const float*)d_in[27];
  const float* spec_ln_b = (const float*)d_in[28];
  const float* spec_W1   = (const float*)d_in[29];
  const float* spec_b1   = (const float*)d_in[30];
  const float* spec_W2   = (const float*)d_in[31];
  const float* spec_b2   = (const float*)d_in[32];
  const float* conf_ln_g = (const float*)d_in[33];
  const float* conf_ln_b = (const float*)d_in[34];
  const float* conf_W1   = (const float*)d_in[35];
  const float* conf_b1   = (const float*)d_in[36];
  const float* conf_W2   = (const float*)d_in[37];
  const float* conf_b2   = (const float*)d_in[38];

  char* ws = (char*)d_ws;
  float*  bps   = (float*)(ws + O_BPS);
  __bf16* pWb   = (__bf16*)(ws + O_PWB);
  __bf16* gW1b  = (__bf16*)(ws + O_GW1B);
  __bf16* gW2b  = (__bf16*)(ws + O_GW2B);
  __bf16* bW1b  = (__bf16*)(ws + O_BW1B);
  __bf16* bW2b  = (__bf16*)(ws + O_BW2B);
  float*  xinF  = (float*)(ws + O_XINF);
  float*  gF    = (float*)(ws + O_GF);
  float*  hF    = (float*)(ws + O_HF);
  __bf16* xBF   = (__bf16*)(ws + O_XBF);
  __bf16* hidBF = (__bf16*)(ws + O_HIDBF);

  // prep: breakpoints + bf16 weight conversion
  bps_kernel<<<2, 256, 0, stream>>>(bp_min, bp_dr, bps);
  cvt_kernel<<<1024, 256, 0, stream>>>(proj_W, pWb, 2048 * 512);
  cvt_kernel<<<512, 256, 0, stream>>>(gate_W1, gW1b, 512 * 256);
  cvt_kernel<<<512, 256, 0, stream>>>(gate_W2, gW2b, 256 * 512);
  cvt_kernel<<<512, 256, 0, stream>>>(bb_W1, bW1b, 512 * 256);
  cvt_kernel<<<256, 256, 0, stream>>>(bb_W2, bW2b, 256 * 256);

  // fused embedding + projection (x_in)
  emb_proj_kernel<<<dim3(B_ / BM, D_ / BN), 256, 0, stream>>>(
      x_num, basis_w, basis_b, bps, pWb, proj_b, xinF, xBF);
  // gate MLP: 512 -> 256 (relu, bf16) -> 512 (sigmoid; emit g f32 and x_gated bf16)
  gemm_bf16_kernel<0><<<dim3(B_ / BM, GH_ / BN), 256, 0, stream>>>(
      xBF, gW1b, gate_b1, nullptr, nullptr, hidBF, GH_, D_);
  gemm_bf16_kernel<1><<<dim3(B_ / BM, D_ / BN), 256, 0, stream>>>(
      hidBF, gW2b, gate_b2, xinF, gF, xBF, D_, GH_);
  // backbone: 512 -> 256 (relu, bf16) -> 256 (relu, f32 h)
  gemm_bf16_kernel<0><<<dim3(B_ / BM, DT_ / BN), 256, 0, stream>>>(
      xBF, bW1b, bb_b1, nullptr, nullptr, hidBF, DT_, D_);
  gemm_bf16_kernel<2><<<dim3(B_ / BM, DT_ / BN), 256, 0, stream>>>(
      hidBF, bW2b, bb_b2, nullptr, hF, nullptr, DT_, DT_);

  // heads: wave per row
  heads_kernel<<<B_ / 8, 256, 0, stream>>>(
      gF, xinF, hF, base_W, base_b,
      safe_ln_g, safe_ln_b, safe_W1, safe_b1, safe_W2, safe_b2,
      top_W1, top_b1, top_W2, top_b2,
      spec_ln_g, spec_ln_b, spec_W1, spec_b1, spec_W2, spec_b2,
      conf_ln_g, conf_ln_b, conf_W1, conf_b1, conf_W2, conf_b2,
      (float*)d_out);

  (void)in_sizes; (void)n_in; (void)out_size; (void)ws_size;
}
